// GatDiffPool_18433999634739
// MI455X (gfx1250) — compile-verified, matrix operand order checked by reading, not run on hardware
//
#include <hip/hip_runtime.h>
#include <hip/hip_bf16.h>
#include <stdint.h>

#define CDIV(a, b) (((a) + (b) - 1) / (b))

// ---------------------------------------------------------------------------
// CDNA5 WMMA types
// ---------------------------------------------------------------------------
typedef __bf16 v16bf __attribute__((ext_vector_type(16)));
typedef __bf16 v8bf  __attribute__((ext_vector_type(8)));
typedef float  v8f   __attribute__((ext_vector_type(8)));

__device__ __forceinline__ unsigned bfbits(float f) {
  // round-to-nearest-even f32 -> bf16 bits
  unsigned u = __float_as_uint(f);
  return (u + 0x7FFFu + ((u >> 16) & 1u)) >> 16;
}
__device__ __forceinline__ __bf16 f2bf(float f) {
  unsigned short h = (unsigned short)bfbits(f);
  return __builtin_bit_cast(__bf16, h);
}
__device__ __forceinline__ unsigned pk2bf(float lo, float hi) {
  return bfbits(lo) | (bfbits(hi) << 16);
}

// ordered encoding so unsigned atomicMax == float max
__device__ __forceinline__ unsigned fenc(float f) {
  unsigned u = __float_as_uint(f);
  return (u & 0x80000000u) ? ~u : (u | 0x80000000u);
}
__device__ __forceinline__ float fdec(unsigned u) {
  u = (u & 0x80000000u) ? (u & 0x7fffffffu) : ~u;
  return __uint_as_float(u);
}

__device__ __forceinline__ v16bf cat8(v8bf lo, v8bf hi) {
  return __builtin_shufflevector(lo, hi, 0, 1, 2, 3, 4, 5, 6, 7,
                                 8, 9, 10, 11, 12, 13, 14, 15);
}

// ---------------------------------------------------------------------------
// Batched/strided GEMM, EXACT dims: M % 64 == 0, N % 32 == 0, K % 32 == 0.
//   C[b] = A[b](MxK) * B[b](KxN) (+bias[n], optional relu)
//   A(m,k) = A[b*sAb + m*sAm + k*sAk],  B(k,n) = B[b*sBb + k*sBk + n*sBn]
// Compile-time layout specialization: AK <=> sAk==1, BK <=> sBk==1.
// 128 threads = 4 waves. Block tile 64(M) x 32(N); each wave owns 16 rows and
// both 16-col halves -> 2 x v_wmma_f32_16x16x32_bf16 per k-step.
// A tile is [m][k]; B tile stored TRANSPOSED ([n][k]) so both fragments are
// two contiguous 16B runs per lane -> ds_load_b128. Row pitch 40 halfs (80B).
// Staging: per-thread base pointers + invariant byte steps; loads batched into
// registers (one wait per tile); k-contig operands use float4 global loads +
// packed uint2 (2xbf16) LDS stores.
// ---------------------------------------------------------------------------
template <bool AK, bool BK>
__global__ void __launch_bounds__(128)
gemm_wmma_bf16(const float* __restrict__ A, long long sAm, long long sAk, long long sAb,
               const float* __restrict__ B, long long sBk, long long sBn, long long sBb,
               float* __restrict__ C, long long ldc, long long sCb,
               const float* __restrict__ bias,
               int N, int K, int relu)
{
  __shared__ __attribute__((aligned(16))) __bf16 As[64][40];   // [m][k]
  __shared__ __attribute__((aligned(16))) __bf16 BsT[32][40];  // [n][k]
  const int tid  = threadIdx.x;
  const int bz   = blockIdx.z;
  const int m0   = blockIdx.y * 64;
  const int n0   = blockIdx.x * 32;
  const int w    = tid >> 5;
  const int lane = tid & 31;
  const int half = lane >> 4;
  const int rc   = lane & 15;
  const int kb   = half * 8;          // 0 or 8
  const int arow = w * 16 + rc;       // wave w owns rows [w*16, w*16+16)
  const float* Ab = A + (size_t)bz * sAb;
  const float* Bb = B + (size_t)bz * sBb;
  (void)sAk; (void)sBk;

  v8f acc0 = {}, acc1 = {};
  for (int kc = 0; kc < K; kc += 32) {
    // ---------------- stage A (64 x 32) ----------------
    if constexpr (AK) {
      // k-contiguous: float4 loads, packed bf16x2 stores (4 quads/thread)
      const int m_l = tid >> 3;            // 0..15 (advance by 16/iter)
      const int ke  = (tid & 7) * 4;       // k element offset of quad
      const char* p = (const char*)(Ab + (long long)(m0 + m_l) * sAm + kc + ke);
      const long long pstep = sAm * 64;    // 16 rows * 4B
      float4 t[4];
#pragma unroll
      for (int it = 0; it < 4; ++it) { t[it] = *(const float4*)p; p += pstep; }
      uint2* q = (uint2*)&As[m_l][ke];     // row pitch = 10 uint2
#pragma unroll
      for (int it = 0; it < 4; ++it) {
        uint2 u; u.x = pk2bf(t[it].x, t[it].y); u.y = pk2bf(t[it].z, t[it].w);
        q[it * 16 * 10] = u;
      }
    } else {
      // m-contiguous (transposed A): scalar loads coalesced over m
      const float* p = Ab + (long long)(m0 + (tid & 63)) * sAm +
                       (long long)(kc + (tid >> 6)) * sAk;
      const long long pstep = sAk * 2;     // elements
      float t[16];
#pragma unroll
      for (int it = 0; it < 16; ++it) { t[it] = p[0]; p += pstep; }
      __bf16* q = &As[tid & 63][tid >> 6];
#pragma unroll
      for (int it = 0; it < 16; ++it) q[it * 2] = f2bf(t[it]);
    }
    // ---------------- stage B transposed into [n][k] (32 x 32) ----------------
    if constexpr (BK) {
      const int n_l = tid >> 3;            // 0..15 (advance by 16/iter)
      const int ke  = (tid & 7) * 4;
      const char* p = (const char*)(Bb + (long long)(n0 + n_l) * sBn + kc + ke);
      const long long pstep = sBn * 64;
      float4 t[2];
#pragma unroll
      for (int it = 0; it < 2; ++it) { t[it] = *(const float4*)p; p += pstep; }
      uint2* q = (uint2*)&BsT[n_l][ke];
#pragma unroll
      for (int it = 0; it < 2; ++it) {
        uint2 u; u.x = pk2bf(t[it].x, t[it].y); u.y = pk2bf(t[it].z, t[it].w);
        q[it * 16 * 10] = u;
      }
    } else {
      // n-contiguous B (e.g. adj): scalar loads coalesced over n
      const float* p = Bb + (long long)(kc + (tid >> 5)) * sBk +
                       (long long)(n0 + (tid & 31)) * sBn;
      const long long pstep = sBk * 4;
      float t[8];
#pragma unroll
      for (int it = 0; it < 8; ++it) { t[it] = p[0]; p += pstep; }
      __bf16* q = &BsT[tid & 31][tid >> 5];
#pragma unroll
      for (int it = 0; it < 8; ++it) q[it * 4] = f2bf(t[it]);
    }
    __syncthreads();

    // ---- fragments: two 16B contiguous runs each -> ds_load_b128 x2 ----
    v16bf af = cat8(*(const v8bf*)&As[arow][kb],
                    *(const v8bf*)&As[arow][16 + kb]);
    v8bf b00 = *(const v8bf*)&BsT[rc][kb];        // cols n0..n0+15
    v8bf b01 = *(const v8bf*)&BsT[rc][16 + kb];
    v8bf b10 = *(const v8bf*)&BsT[16 + rc][kb];   // cols n0+16..n0+31
    v8bf b11 = *(const v8bf*)&BsT[16 + rc][16 + kb];
    acc0 = __builtin_amdgcn_wmma_f32_16x16x32_bf16(false, af, false, cat8(b00, b01),
                                                   (short)0, acc0, false, false);
    acc1 = __builtin_amdgcn_wmma_f32_16x16x32_bf16(false, af, false, cat8(b10, b11),
                                                   (short)0, acc1, false, false);
    __syncthreads();
  }

  float* Cb = C + (size_t)bz * sCb;
#pragma unroll
  for (int r = 0; r < 8; ++r) {
    int gm = m0 + w * 16 + r + 8 * half;
    int gn0 = n0 + rc;
    int gn1 = n0 + 16 + rc;
    float v0 = acc0[r], v1 = acc1[r];
    if (bias) { v0 += bias[gn0]; v1 += bias[gn1]; }
    if (relu) { v0 = fmaxf(v0, 0.f); v1 = fmaxf(v1, 0.f); }
    Cb[(long long)gm * ldc + gn0] = v0;
    Cb[(long long)gm * ldc + gn1] = v1;
  }
  (void)N;
}

// host-side dispatch on operand layout (compile-time specialization)
static void launch_gemm(hipStream_t st,
                        const float* A, long long sAm, long long sAk, long long sAb,
                        const float* B, long long sBk, long long sBn, long long sBb,
                        float* C, long long ldc, long long sCb, const float* bias,
                        int M, int N, int K, int relu, int batches)
{
  dim3 grid(CDIV(N, 32), CDIV(M, 64), batches);
  if (sAk == 1 && sBk == 1)
    gemm_wmma_bf16<true, true><<<grid, 128, 0, st>>>(A, sAm, sAk, sAb, B, sBk, sBn, sBb,
                                                     C, ldc, sCb, bias, N, K, relu);
  else if (sAk == 1)
    gemm_wmma_bf16<true, false><<<grid, 128, 0, st>>>(A, sAm, sAk, sAb, B, sBk, sBn, sBb,
                                                      C, ldc, sCb, bias, N, K, relu);
  else if (sBk == 1)
    gemm_wmma_bf16<false, true><<<grid, 128, 0, st>>>(A, sAm, sAk, sAb, B, sBk, sBn, sBb,
                                                      C, ldc, sCb, bias, N, K, relu);
  else
    gemm_wmma_bf16<false, false><<<grid, 128, 0, st>>>(A, sAm, sAk, sAb, B, sBk, sBn, sBb,
                                                       C, ldc, sCb, bias, N, K, relu);
}

// ---------------------------------------------------------------------------
// Fused link loss:  acc += sum over tile of (adj[b,m,n] - (S S^T)[b,m,n])^2
// S is [B, Nn, Cc] row-major; Nn % 32 == 0, Cc % 32 == 0 (exact dims).
// 32x32 block tile, 4 waves each owning one 16x16 sub-tile. Both operands are
// k-contiguous -> vectorized staging.
// ---------------------------------------------------------------------------
__global__ void __launch_bounds__(128)
link_loss_wmma(const float* __restrict__ S, const float* __restrict__ adj,
               int Nn, int Cc, float* __restrict__ acc_out)
{
  __shared__ __attribute__((aligned(16))) __bf16 As[32][40];   // [m][k]
  __shared__ __attribute__((aligned(16))) __bf16 BsT[32][40];  // [n][k]
  __shared__ float red[128];
  const int tid  = threadIdx.x;
  const int bz   = blockIdx.z;
  const int m0   = blockIdx.y * 32;
  const int n0   = blockIdx.x * 32;
  const int w    = tid >> 5, lane = tid & 31;
  const int wm = (w >> 1) * 16, wn = (w & 1) * 16;
  const int half = lane >> 4, rc = lane & 15, kb = half * 8;
  const float* Sb   = S   + (size_t)bz * Nn * Cc;
  const float* adjb = adj + (size_t)bz * Nn * Nn;

  v8f acc = {};
  for (int kc = 0; kc < Cc; kc += 32) {
    const int r_l = tid >> 3;             // 0..15 (advance by 16/iter)
    const int ke  = (tid & 7) * 4;
    const char* pa = (const char*)(Sb + (long long)(m0 + r_l) * Cc + kc + ke);
    const char* pb = (const char*)(Sb + (long long)(n0 + r_l) * Cc + kc + ke);
    const long long pstep = (long long)Cc * 64;
    float4 ta[2], tb[2];
#pragma unroll
    for (int it = 0; it < 2; ++it) {
      ta[it] = *(const float4*)pa; pa += pstep;
      tb[it] = *(const float4*)pb; pb += pstep;
    }
    uint2* qa = (uint2*)&As[r_l][ke];
    uint2* qb = (uint2*)&BsT[r_l][ke];
#pragma unroll
    for (int it = 0; it < 2; ++it) {
      uint2 ua; ua.x = pk2bf(ta[it].x, ta[it].y); ua.y = pk2bf(ta[it].z, ta[it].w);
      uint2 ub; ub.x = pk2bf(tb[it].x, tb[it].y); ub.y = pk2bf(tb[it].z, tb[it].w);
      qa[it * 16 * 10] = ua;
      qb[it * 16 * 10] = ub;
    }
    __syncthreads();
    v16bf af  = cat8(*(const v8bf*)&As[wm + rc][kb],
                     *(const v8bf*)&As[wm + rc][16 + kb]);
    v16bf bfv = cat8(*(const v8bf*)&BsT[wn + rc][kb],
                     *(const v8bf*)&BsT[wn + rc][16 + kb]);
    acc = __builtin_amdgcn_wmma_f32_16x16x32_bf16(false, af, false, bfv,
                                                  (short)0, acc, false, false);
    __syncthreads();
  }

  float part = 0.f;
#pragma unroll
  for (int r = 0; r < 8; ++r) {
    int gm = m0 + wm + r + 8 * half;
    int gn = n0 + wn + rc;
    float d = adjb[(size_t)gm * Nn + gn] - acc[r];
    part += d * d;
  }
  red[tid] = part;
  __syncthreads();
  for (int s = 64; s > 0; s >>= 1) { if (tid < s) red[tid] += red[tid + s]; __syncthreads(); }
  if (tid == 0) atomicAdd(acc_out, red[0]);
}

// ---------------------------------------------------------------------------
// GATv2 irregular kernels (wave-per-edge gathers, wave32 shfl reductions)
// ---------------------------------------------------------------------------
__global__ void gat_edge_scores(const float* __restrict__ xl, const float* __restrict__ xr,
                                const float* __restrict__ att,
                                const int* __restrict__ src, const int* __restrict__ dst,
                                int E0, int NT, int C, float* __restrict__ e_out)
{
  int widx = blockIdx.x * (blockDim.x >> 5) + (threadIdx.x >> 5);
  int lane = threadIdx.x & 31;
  int Etot = E0 + NT;
  if (widx >= Etot) return;
  int s, d;
  if (widx < E0) { s = src[widx]; d = dst[widx]; } else { s = d = widx - E0; }
  __builtin_prefetch(xl + (size_t)s * C + lane * 4, 0, 1);   // global_prefetch_b8
  __builtin_prefetch(xr + (size_t)d * C + lane * 4, 0, 1);
  float sum = 0.f;
  for (int c = lane; c < C; c += 32) {
    float v = xl[(size_t)s * C + c] + xr[(size_t)d * C + c];
    v = v > 0.f ? v : 0.2f * v;                               // leaky_relu(0.2)
    sum += v * att[c];
  }
  for (int o = 16; o; o >>= 1) sum += __shfl_xor(sum, o, 32);
  if (lane == 0) e_out[widx] = sum;
}

__global__ void seg_max_kernel(const float* __restrict__ e, const int* __restrict__ dst,
                               int E0, int NT, unsigned* __restrict__ emax_u)
{
  int i = blockIdx.x * blockDim.x + threadIdx.x;
  int Etot = E0 + NT;
  if (i >= Etot) return;
  int d = (i < E0) ? dst[i] : (i - E0);
  atomicMax(&emax_u[d], fenc(e[i]));
}

__global__ void seg_expsum_kernel(float* __restrict__ e, const int* __restrict__ dst,
                                  int E0, int NT, const unsigned* __restrict__ emax_u,
                                  float* __restrict__ den)
{
  int i = blockIdx.x * blockDim.x + threadIdx.x;
  int Etot = E0 + NT;
  if (i >= Etot) return;
  int d = (i < E0) ? dst[i] : (i - E0);
  float ez = __expf(e[i] - fdec(emax_u[d]));
  e[i] = ez;
  atomicAdd(&den[d], ez);
}

__global__ void gat_scatter(const float* __restrict__ xl, const float* __restrict__ ez,
                            const float* __restrict__ den,
                            const int* __restrict__ src, const int* __restrict__ dst,
                            int E0, int NT, int C, float* __restrict__ out)
{
  int widx = blockIdx.x * (blockDim.x >> 5) + (threadIdx.x >> 5);
  int lane = threadIdx.x & 31;
  int Etot = E0 + NT;
  if (widx >= Etot) return;
  int s, d;
  if (widx < E0) { s = src[widx]; d = dst[widx]; } else { s = d = widx - E0; }
  float w = ez[widx] / den[d];
  __builtin_prefetch(xl + (size_t)s * C + lane * 4, 0, 1);
  for (int c = lane; c < C; c += 32)
    atomicAdd(&out[(size_t)d * C + c], w * xl[(size_t)s * C + c]);
}

// ---------------------------------------------------------------------------
// Pointwise / reduction helpers
// ---------------------------------------------------------------------------
__global__ void fill_f32(float* p, float v, int n) {
  int i = blockIdx.x * blockDim.x + threadIdx.x;
  if (i < n) p[i] = v;
}
__global__ void fill_u32(unsigned* p, unsigned v, int n) {
  int i = blockIdx.x * blockDim.x + threadIdx.x;
  if (i < n) p[i] = v;
}
__global__ void init_rows_bias(float* __restrict__ out, const float* __restrict__ bias,
                               int NT, int C) {
  int i = blockIdx.x * blockDim.x + threadIdx.x;
  if (i < NT * C) out[i] = bias[i % C];
}
__global__ void iota_grid(int* __restrict__ rows, int* __restrict__ cols, int n, int C0) {
  int i = blockIdx.x * blockDim.x + threadIdx.x;
  if (i < n) { rows[i] = i / C0; cols[i] = i % C0; }
}
__global__ void adj_build(const int* __restrict__ src, const int* __restrict__ dst,
                          const int* __restrict__ batch, float* __restrict__ adj,
                          int E0, int N) {
  int i = blockIdx.x * blockDim.x + threadIdx.x;
  if (i >= E0) return;
  int s = src[i], d = dst[i];
  int g = batch[s];
  atomicAdd(&adj[(size_t)g * N * N + (size_t)(s - g * N) * N + (d - g * N)], 1.f);
}

// layernorm(+relu)(+skip); blockDim == C (128)
__global__ void ln_relu_kernel(const float* __restrict__ in, const float* __restrict__ g,
                               const float* __restrict__ bb, const float* __restrict__ skip,
                               float* __restrict__ out, int C)
{
  __shared__ float red[128];
  int n = blockIdx.x, c = threadIdx.x;
  float v = in[(size_t)n * C + c];
  red[c] = v; __syncthreads();
  for (int s = C >> 1; s > 0; s >>= 1) { if (c < s) red[c] += red[c + s]; __syncthreads(); }
  float mean = red[0] / C; __syncthreads();
  float dv = v - mean;
  red[c] = dv * dv; __syncthreads();
  for (int s = C >> 1; s > 0; s >>= 1) { if (c < s) red[c] += red[c + s]; __syncthreads(); }
  float var = red[0] / C;
  float y = dv * rsqrtf(var + 1e-5f) * g[c] + bb[c];
  y = fmaxf(y, 0.f);
  if (skip) y += skip[(size_t)n * C + c];
  out[(size_t)n * C + c] = y;
}

// row softmax over C columns + entropy accumulation; blockDim == C (256)
__global__ void softmax_ent_kernel(float* __restrict__ s, int C, float* __restrict__ ent_acc)
{
  __shared__ float red[256];
  int n = blockIdx.x, c = threadIdx.x;
  float v = s[(size_t)n * C + c];
  red[c] = v; __syncthreads();
  for (int st = C >> 1; st > 0; st >>= 1) { if (c < st) red[c] = fmaxf(red[c], red[c + st]); __syncthreads(); }
  float m = red[0]; __syncthreads();
  float ez = __expf(v - m);
  red[c] = ez; __syncthreads();
  for (int st = C >> 1; st > 0; st >>= 1) { if (c < st) red[c] += red[c + st]; __syncthreads(); }
  float den = red[0]; __syncthreads();
  float p = ez / den;
  s[(size_t)n * C + c] = p;
  red[c] = -p * __logf(p + 1e-15f); __syncthreads();
  for (int st = C >> 1; st > 0; st >>= 1) { if (c < st) red[c] += red[c + st]; __syncthreads(); }
  if (c == 0) atomicAdd(ent_acc, red[0]);
}

// stage-2 pool (softmax over 1 cluster == all-ones): out[b,c] = sum_n xf[b,n,c]
__global__ void pool_sum_kernel(const float* __restrict__ xf, float* __restrict__ dst,
                                int Cn, int C, int ldd)
{
  int b = blockIdx.x, c = threadIdx.x;
  float s = 0.f;
  for (int nn = 0; nn < Cn; ++nn) s += xf[((size_t)b * Cn + nn) * C + c];
  dst[(size_t)b * ldd + c] = s;
}

// stage-2 link loss: sum (adj2 - 1)^2   (S S^T with all-ones S is the 1-matrix)
__global__ void link2_kernel(const float* __restrict__ a, int n, float* __restrict__ acc)
{
  float s = 0.f;
  for (int i = blockIdx.x * blockDim.x + threadIdx.x; i < n; i += gridDim.x * blockDim.x) {
    float d = a[i] - 1.f; s += d * d;
  }
  for (int o = 16; o; o >>= 1) s += __shfl_xor(s, o, 32);
  if ((threadIdx.x & 31) == 0) atomicAdd(acc, s);
}

__global__ void final_linear(const float* __restrict__ xcat, const float* __restrict__ W,
                             const float* __restrict__ bias, float* __restrict__ out,
                             int Bn, int Cin, int Cout)
{
  int i = blockIdx.x * blockDim.x + threadIdx.x;
  if (i >= Bn * Cout) return;
  int b = i / Cout, o = i % Cout;
  float s = bias[o];
  for (int c = 0; c < Cin; ++c) s += xcat[(size_t)b * Cin + c] * W[(size_t)o * Cin + c];
  out[i] = fmaxf(s, 0.f);
}

__global__ void pack_losses(const float* __restrict__ acc, float* __restrict__ out)
{
  if (threadIdx.x == 0) {
    const float adjsz0 = 4.f * 512.f * 512.f;
    const float adjsz1 = 4.f * 256.f * 256.f;
    out[256 + 0] = sqrtf(acc[0]) / adjsz0;   // link g0 k0
    out[256 + 1] = sqrtf(acc[1]) / adjsz1;   // link g0 k1
    out[256 + 2] = sqrtf(acc[2]) / adjsz0;   // link g1 k0
    out[256 + 3] = sqrtf(acc[3]) / adjsz1;   // link g1 k1
    out[260 + 0] = acc[4] / 2048.f;          // ent g0 k0
    out[260 + 1] = 0.f;                      // ent g0 k1 (softmax over 1 cluster)
    out[260 + 2] = acc[6] / 2048.f;          // ent g1 k0
    out[260 + 3] = 0.f;                      // ent g1 k1
  }
}

// ---------------------------------------------------------------------------
// Host-side GATv2 layer driver (all launches on `st`)
// ---------------------------------------------------------------------------
static void run_gat_layer(hipStream_t st, const float* xin, int NT, int Cin,
                          const float* Wl, const float* bl,
                          const float* Wr, const float* br,
                          const float* att, const float* bias,
                          const int* src, const int* dst, int E0,
                          float* XL, float* XR, float* ESC, float* DEN,
                          unsigned* EMAXU, float* Hout, int Cout)
{
  // xl = x @ Wl^T + bl ; xr = x @ Wr^T + br   (B(k,n) = W[n*Cin+k])
  launch_gemm(st, xin, Cin, 1, 0, Wl, 1, Cin, 0, XL, Cout, 0, bl, NT, Cout, Cin, 0, 1);
  launch_gemm(st, xin, Cin, 1, 0, Wr, 1, Cin, 0, XR, Cout, 0, br, NT, Cout, Cin, 0, 1);
  int Etot = E0 + NT;
  gat_edge_scores<<<CDIV(Etot, 8), 256, 0, st>>>(XL, XR, att, src, dst, E0, NT, Cout, ESC);
  fill_u32<<<CDIV(NT, 256), 256, 0, st>>>(EMAXU, 0x007FFFFFu /* fenc(-inf) */, NT);
  fill_f32<<<CDIV(NT, 256), 256, 0, st>>>(DEN, 0.f, NT);
  seg_max_kernel<<<CDIV(Etot, 256), 256, 0, st>>>(ESC, dst, E0, NT, EMAXU);
  seg_expsum_kernel<<<CDIV(Etot, 256), 256, 0, st>>>(ESC, dst, E0, NT, EMAXU, DEN);
  init_rows_bias<<<CDIV(NT * Cout, 256), 256, 0, st>>>(Hout, bias, NT, Cout);
  gat_scatter<<<CDIV(Etot, 8), 256, 0, st>>>(XL, ESC, DEN, src, dst, E0, NT, Cout, Hout);
}

// ---------------------------------------------------------------------------
// Entry point.
// Input layout (jax pytree flatten order: dict keys sorted):
//  [0]=x [1]=edge_indices [2]=batch [3]=final.W [4]=final.b
//  then per graph i (36 leaves, base = 5+36*i):
//   init_gat[0]: +0 Wl +1 Wr +2 att +3 bias +4 bl +5 br
//   init_gat[1]: +6..+11 (same order)
//   init_norm[0]: +12 b +13 g ; init_norm[1]: +14 b +15 g
//   pool_lin[0]: +16 W +17 b ; pool_lin[1]: +18 W +19 b
//   post_gat[0][0]: +20..+25 ; post_gat[0][1]: +26..+31
//   post_norm[0][0]: +32 b +33 g ; post_norm[0][1]: +34 b +35 g
// Output: 264 floats = relu-out[4,64] ++ link_losses[4] ++ ent_losses[4]
// ---------------------------------------------------------------------------
extern "C" void kernel_launch(void* const* d_in, const int* in_sizes, int n_in,
                              void* d_out, int out_size, void* d_ws, size_t ws_size,
                              hipStream_t stream)
{
  (void)n_in; (void)out_size; (void)ws_size;
  const float* x     = (const float*)d_in[0];
  const int*   ei    = (const int*)d_in[1];
  const int*   batch = (const int*)d_in[2];
  const float* Wf    = (const float*)d_in[3];
  const float* bfin  = (const float*)d_in[4];

  const int NT = 2048, Bn = 4, Nn = 512, C0 = 256, HIDc = 128, INC = 64;
  const int E  = in_sizes[1] / 4;            // edges per edge-set (131072)

  // ---- workspace carve-up (floats, 64-elem aligned) ----
  float* fw = (float*)d_ws;
  size_t off = 0;
  auto alloc = [&](size_t nelem) {
    float* p = fw + off;
    off += (nelem + 63) & ~(size_t)63;
    return p;
  };
  float*    XL    = alloc((size_t)NT * HIDc);
  float*    XR    = alloc((size_t)NT * HIDc);
  float*    H     = alloc((size_t)NT * HIDc);
  float*    XG    = alloc((size_t)NT * HIDc);
  float*    ESC   = alloc(266240);                      // >= 262144 + 1024 scores
  float*    DEN   = alloc(NT);
  float*    ADJ   = alloc((size_t)Bn * Nn * Nn);        // 4x512x512
  float*    S     = alloc((size_t)Bn * Nn * C0);        // 4x512x256
  float*    XNEW  = alloc((size_t)Bn * C0 * HIDc);      // 4x256x128 (also xf)
  float*    T     = alloc((size_t)Bn * C0 * Nn);        // 4x256x512
  float*    ADJ2  = alloc((size_t)Bn * C0 * C0);        // 4x256x256
  float*    XCAT  = alloc((size_t)Bn * 2 * HIDc);       // [4, 256]
  float*    ACC   = alloc(64);                          // [0..3] link, [4..7] ent
  unsigned* EMAXU = (unsigned*)alloc(NT);
  int*      ROWS  = (int*)alloc((size_t)Bn * C0 * C0);  // 262144
  int*      COLS  = (int*)alloc((size_t)Bn * C0 * C0);

  fill_f32<<<1, 64, 0, stream>>>(ACC, 0.f, 16);
  // dense post-pool edge grid: rows = repeat(0..1023, 256), cols = tile(0..255)
  const int NTp = Bn * C0;        // 1024 dense nodes
  const int Ep  = NTp * C0;       // 262144 dense edges
  iota_grid<<<CDIV(Ep, 256), 256, 0, stream>>>(ROWS, COLS, Ep, C0);

  for (int gi = 0; gi < 2; ++gi) {
    const int base = 5 + gi * 36;
    auto P = [&](int k) { return (const float*)d_in[base + k]; };
    const int* src = ei + (size_t)gi * 2 * E;
    const int* dst = src + E;

    // ---- init GAT stack (sparse edges) ----
    run_gat_layer(stream, x, NT, INC, P(0), P(4), P(1), P(5), P(2), P(3),
                  src, dst, E, XL, XR, ESC, DEN, EMAXU, H, HIDc);
    ln_relu_kernel<<<NT, HIDc, 0, stream>>>(H, P(13), P(12), nullptr, XG, HIDc);
    run_gat_layer(stream, XG, NT, HIDc, P(6), P(10), P(7), P(11), P(8), P(9),
                  src, dst, E, XL, XR, ESC, DEN, EMAXU, H, HIDc);
    ln_relu_kernel<<<NT, HIDc, 0, stream>>>(H, P(15), P(14), XG, XG, HIDc);

    // ---- dense adjacency from edges ----
    fill_f32<<<CDIV(Bn * Nn * Nn, 256), 256, 0, stream>>>(ADJ, 0.f, Bn * Nn * Nn);
    adj_build<<<CDIV(E, 256), 256, 0, stream>>>(src, dst, batch, ADJ, E, Nn);

    // ---- diff-pool stage 0: assignments + softmax + entropy ----
    launch_gemm(stream, XG, HIDc, 1, 0, P(16), 1, HIDc, 0,
                S, C0, 0, P(17), NT, C0, HIDc, 0, 1);
    softmax_ent_kernel<<<NT, C0, 0, stream>>>(S, C0, ACC + 4 + gi * 2);

    // x_new = S^T X : [B,256,128]
    launch_gemm(stream, S, 1, C0, (long long)Nn * C0, XG, HIDc, 1, (long long)Nn * HIDc,
                XNEW, HIDc, (long long)C0 * HIDc, nullptr, C0, HIDc, Nn, 0, Bn);
    // T = S^T A : [B,256,512]
    launch_gemm(stream, S, 1, C0, (long long)Nn * C0, ADJ, Nn, 1, (long long)Nn * Nn,
                T, Nn, (long long)C0 * Nn, nullptr, C0, Nn, Nn, 0, Bn);
    // adj_new = T S : [B,256,256]
    launch_gemm(stream, T, Nn, 1, (long long)C0 * Nn, S, C0, 1, (long long)Nn * C0,
                ADJ2, C0, (long long)C0 * C0, nullptr, C0, C0, Nn, 0, Bn);
    // link loss 0 (fused S S^T vs adj)
    link_loss_wmma<<<dim3(Nn / 32, Nn / 32, Bn), 128, 0, stream>>>(
        S, ADJ, Nn, C0, ACC + gi * 2);

    // ---- post-pool GAT stack on dense grid (xf = XNEW as [1024,128]) ----
    for (int l = 0; l < 2; ++l) {
      int pb = 20 + l * 6;
      run_gat_layer(stream, XNEW, NTp, HIDc, P(pb + 0), P(pb + 4), P(pb + 1),
                    P(pb + 5), P(pb + 2), P(pb + 3),
                    ROWS, COLS, Ep, XL, XR, ESC, DEN, EMAXU, H, HIDc);
      ln_relu_kernel<<<NTp, HIDc, 0, stream>>>(H, P(33 + l * 2), P(32 + l * 2),
                                               XNEW, XNEW, HIDc);
    }

    // ---- diff-pool stage 1 (1 cluster: softmax == ones) ----
    pool_sum_kernel<<<Bn, HIDc, 0, stream>>>(XNEW, XCAT + gi * HIDc, C0, HIDc, 2 * HIDc);
    link2_kernel<<<64, 256, 0, stream>>>(ADJ2, Bn * C0 * C0, ACC + gi * 2 + 1);
  }

  final_linear<<<1, 256, 0, stream>>>(XCAT, Wf, bfin, (float*)d_out, Bn, 2 * HIDc, 64);
  pack_losses<<<1, 32, 0, stream>>>(ACC, (float*)d_out);
}
